// Attention_11158325035119
// MI455X (gfx1250) — matrix operand order repaired and checked by reading, compiled, not run on hardware
//
#include <hip/hip_runtime.h>

// ---------------------------------------------------------------------------
// CDNA5 (gfx1250) attention block: LN -> QKV proj -> flash attention -> out proj
// All matmuls via v_wmma_f32_16x16x32_bf16 (wave32, 8 waves/block).
// Tile staging via GLOBAL_LOAD_ASYNC_TO_LDS_B128 (ASYNCcnt), double-buffered
// so DMA overlaps WMMA compute; one barrier per K-step.
// ---------------------------------------------------------------------------

typedef __attribute__((ext_vector_type(16))) __bf16 v16bf;
typedef __attribute__((ext_vector_type(8)))  float  v8f;

union FragAB {
    uint4  u[2];
    v16bf  v;
};

static __device__ __forceinline__ v8f wmma_bf16(const FragAB& a, const FragAB& b, v8f c) {
    return __builtin_amdgcn_wmma_f32_16x16x32_bf16(
        /*neg_a=*/false, a.v, /*neg_b=*/false, b.v,
        /*c_mod=*/(short)0, c, /*reuse_a=*/false, /*reuse_b=*/false);
}

// Async DMA of 16 bytes global -> LDS (GLOBAL_LOAD_ASYNC_TO_LDS_B128).
// LDS address = low 32 bits of the generic shared pointer (aperture rule).
static __device__ __forceinline__ void async_b128(void* lds_dst, const void* gsrc) {
    unsigned lds = (unsigned)(size_t)lds_dst;
    unsigned long long g = (unsigned long long)(size_t)gsrc;
    asm volatile("global_load_async_to_lds_b128 %0, %1, off" :: "v"(lds), "v"(g) : "memory");
}

static __device__ __forceinline__ void wait_async0() {
#if __has_builtin(__builtin_amdgcn_s_wait_asynccnt)
    __builtin_amdgcn_s_wait_asynccnt(0);
#else
    asm volatile("s_wait_asynccnt 0x0" ::: "memory");
#endif
}

// ---------------------------------------------------------------------------
// Transpose + cast: out[n*K + k] = (bf16) in[k*Nc + n].   (weights, once)
// ---------------------------------------------------------------------------
__global__ __launch_bounds__(256) void transpose_cast_bf16(
    const float* __restrict__ in, __bf16* __restrict__ out, int K, int Nc)
{
    __shared__ float t[32][33];
    const int k0 = blockIdx.y * 32, n0 = blockIdx.x * 32;
    const int tx = threadIdx.x & 31, ty = threadIdx.x >> 5;   // 32 x 8
    #pragma unroll
    for (int i = 0; i < 32; i += 8)
        t[ty + i][tx] = in[(size_t)(k0 + ty + i) * Nc + n0 + tx];
    __syncthreads();
    #pragma unroll
    for (int i = 0; i < 32; i += 8)
        out[(size_t)(n0 + ty + i) * K + k0 + tx] = (__bf16)t[tx][ty + i];
}

// ---------------------------------------------------------------------------
// LayerNorm (one row per block, D=1024), bf16 output
// ---------------------------------------------------------------------------
__global__ __launch_bounds__(256) void layernorm_bf16(
    const float* __restrict__ x, const float* __restrict__ g,
    const float* __restrict__ bta, __bf16* __restrict__ xn, int D)
{
    const int row = blockIdx.x;
    const float* xr = x + (size_t)row * D;
    __shared__ float red[256];

    float s = 0.f;
    for (int j = threadIdx.x; j < D; j += 256) s += xr[j];
    red[threadIdx.x] = s; __syncthreads();
    for (int off = 128; off; off >>= 1) {
        if (threadIdx.x < off) red[threadIdx.x] += red[threadIdx.x + off];
        __syncthreads();
    }
    float mu = red[0] / (float)D;
    __syncthreads();

    float v = 0.f;
    for (int j = threadIdx.x; j < D; j += 256) { float d = xr[j] - mu; v += d * d; }
    red[threadIdx.x] = v; __syncthreads();
    for (int off = 128; off; off >>= 1) {
        if (threadIdx.x < off) red[threadIdx.x] += red[threadIdx.x + off];
        __syncthreads();
    }
    float rinv = rsqrtf(red[0] / (float)D + 1e-5f);

    __bf16* o = xn + (size_t)row * D;
    for (int j = threadIdx.x; j < D; j += 256)
        o[j] = (__bf16)((xr[j] - mu) * rinv * g[j] + bta[j]);
}

// ---------------------------------------------------------------------------
// Tiled bf16 GEMM: C[M,Nc] = A[M,K] * Wt^T, with Wt stored [Nc][K] (pre-
// transposed). Block tile 128x128, K-step 32, 8 waves (4 M x 2 N).
// Double-buffered LDS; async DMA for tile t+1 overlaps WMMA on tile t.
// ---------------------------------------------------------------------------
template <bool WRITE_FLOAT>
__global__ __launch_bounds__(256) void gemm_bf16_wmma(
    const __bf16* __restrict__ A, const __bf16* __restrict__ Wt,
    void* __restrict__ Cout, int M, int K, int Nc)
{
    __shared__ __bf16 As[2 * 128 * 40];   // [buf][row][k], stride 40 halves
    __shared__ __bf16 Bs[2 * 128 * 40];   // [buf][col][k]

    const int tid    = threadIdx.x;
    const int lane   = tid & 31;
    const int wave   = tid >> 5;
    const int wm     = wave >> 1;     // 0..3
    const int wn     = wave & 1;      // 0..1
    const int l16    = lane & 15;
    const int half16 = lane >> 4;     // 0 or 1
    const int mblk   = blockIdx.y * 128;
    const int nblk   = blockIdx.x * 128;

    const int ldr0 = tid >> 2;               // rows 0..63
    const int ldk  = (tid & 3) << 3;         // k-chunk 0/8/16/24

    auto issue_tile = [&](int t) {
        __bf16* Ac = As + (t & 1) * (128 * 40);
        __bf16* Bc = Bs + (t & 1) * (128 * 40);
        const size_t k0 = (size_t)t * 32;
        async_b128(&Ac[ldr0 * 40 + ldk],        A  + (size_t)(mblk + ldr0) * K + k0 + ldk);
        async_b128(&Ac[(ldr0 + 64) * 40 + ldk], A  + (size_t)(mblk + ldr0 + 64) * K + k0 + ldk);
        async_b128(&Bc[ldr0 * 40 + ldk],        Wt + (size_t)(nblk + ldr0) * K + k0 + ldk);
        async_b128(&Bc[(ldr0 + 64) * 40 + ldk], Wt + (size_t)(nblk + ldr0 + 64) * K + k0 + ldk);
    };

    const int T = K >> 5;             // number of 32-wide K tiles
    v8f acc[2][4] = {};

    issue_tile(0);

    for (int t = 0; t < T; ++t) {
        wait_async0();                 // tile t DMA complete (this wave)
        __syncthreads();               // all waves: tile t visible; prev compute done
        if (t + 1 < T) issue_tile(t + 1);      // overlaps with compute below
        if (t + 2 < T) {                       // L2 prefetch two tiles ahead
            __builtin_prefetch(A  + (size_t)(mblk + ldr0) * K + (size_t)(t + 2) * 32 + ldk, 0, 1);
            __builtin_prefetch(Wt + (size_t)(nblk + ldr0) * K + (size_t)(t + 2) * 32 + ldk, 0, 1);
        }

        const __bf16* Ac = As + (t & 1) * (128 * 40);
        const __bf16* Bc = Bs + (t & 1) * (128 * 40);

        FragAB afr[2], bfr[4];
        #pragma unroll
        for (int mt = 0; mt < 2; ++mt) {
            int row = wm * 32 + mt * 16 + l16;
            int ko  = half16 * 16;
            afr[mt].u[0] = *(const uint4*)&Ac[row * 40 + ko];
            afr[mt].u[1] = *(const uint4*)&Ac[row * 40 + ko + 8];
        }
        #pragma unroll
        for (int nt = 0; nt < 4; ++nt) {
            int col = wn * 64 + nt * 16 + l16;
            int kb  = half16 * 16;
            bfr[nt].u[0] = *(const uint4*)&Bc[col * 40 + kb];
            bfr[nt].u[1] = *(const uint4*)&Bc[col * 40 + kb + 8];
        }
        #pragma unroll
        for (int mt = 0; mt < 2; ++mt)
            #pragma unroll
            for (int nt = 0; nt < 4; ++nt)
                acc[mt][nt] = wmma_bf16(afr[mt], bfr[nt], acc[mt][nt]);
        // no trailing barrier: next iteration's top barrier orders buffer reuse
    }

    // Epilogue: hoisted base, no branches (template).
    const size_t baseOff = (size_t)(mblk + wm * 32 + half16 * 8) * Nc
                         + (size_t)(nblk + wn * 64 + l16);
    if (WRITE_FLOAT) {
        float* C = (float*)Cout + baseOff;
        #pragma unroll
        for (int mt = 0; mt < 2; ++mt)
            #pragma unroll
            for (int r = 0; r < 8; ++r) {
                float* rp = C + (size_t)(mt * 16 + r) * Nc;
                #pragma unroll
                for (int nt = 0; nt < 4; ++nt) rp[nt * 16] = acc[mt][nt][r];
            }
    } else {
        __bf16* C = (__bf16*)Cout + baseOff;
        #pragma unroll
        for (int mt = 0; mt < 2; ++mt)
            #pragma unroll
            for (int r = 0; r < 8; ++r) {
                __bf16* rp = C + (size_t)(mt * 16 + r) * Nc;
                #pragma unroll
                for (int nt = 0; nt < 4; ++nt) rp[nt * 16] = (__bf16)acc[mt][nt][r];
            }
    }
}

// ---------------------------------------------------------------------------
// Build V^T: vt[((b*16+h)*64 + d)*N + n] = kv[(b*N+n)*2048 + 1024 + h*64 + d]
// ---------------------------------------------------------------------------
__global__ __launch_bounds__(256) void build_vt(
    const __bf16* __restrict__ kv, __bf16* __restrict__ vt, int Nn)
{
    __shared__ __bf16 t[64 * 72];          // t[n][d], stride 72
    const int bh = blockIdx.y;             // b*16 + h
    const int b = bh >> 4, h = bh & 15;
    const int n0 = blockIdx.x * 64;
    const int tx = threadIdx.x & 7;        // d-chunk
    const int ty = threadIdx.x >> 3;       // 0..31

    #pragma unroll
    for (int i = 0; i < 64; i += 32)
        *(uint4*)&t[(ty + i) * 72 + tx * 8] =
            *(const uint4*)(kv + (size_t)(b * Nn + n0 + ty + i) * 2048 + 1024 + h * 64 + tx * 8);
    __syncthreads();

    #pragma unroll
    for (int i = 0; i < 64; i += 32) {
        int d = ty + i;
        union { __bf16 h8[8]; uint4 u; } tmp;
        #pragma unroll
        for (int e = 0; e < 8; ++e) tmp.h8[e] = t[(tx * 8 + e) * 72 + d];
        *(uint4*)&vt[(size_t)(bh * 64 + d) * Nn + n0 + tx * 8] = tmp.u;
    }
}

// ---------------------------------------------------------------------------
// Flash attention: one (b, h, 128-query tile) per block; 8 waves.
// Q fragments hoisted to registers; K/V tiles double-buffered async DMA.
// q: [B*N,1024] bf16; kv: [B*N,2048] bf16 (k|v); vt: [B*H*64, N] bf16.
// ---------------------------------------------------------------------------
__global__ __launch_bounds__(256) void attn_wmma(
    const __bf16* __restrict__ q, const __bf16* __restrict__ kv,
    const __bf16* __restrict__ vt, __bf16* __restrict__ o, int Nn)
{
    extern __shared__ __align__(16) char smem[];
    __bf16* Qs = (__bf16*)smem;               // [128][72]
    __bf16* Ks = Qs + 128 * 72;               // [2][128][72]
    __bf16* Vs = Ks + 2 * 128 * 72;           // [2][64][136]  ([dh][key])
    __bf16* Ps = Vs + 2 * 64 * 136;           // [128][136]
    float* mrun = (float*)(Ps + 128 * 136);   // [128]
    float* lrun = mrun + 128;                 // [128]
    float* scl  = lrun + 128;                 // [128]
    float* pmax = scl + 128;                  // [2][128]
    float* psum = pmax + 256;                 // [2][128]

    const int tid    = threadIdx.x;
    const int lane   = tid & 31;
    const int wave   = tid >> 5;
    const int wm     = wave >> 1;
    const int wn     = wave & 1;
    const int l16    = lane & 15;
    const int half16 = lane >> 4;
    const int qt = blockIdx.x, h = blockIdx.y, b = blockIdx.z;
    const int bh = b * 16 + h;
    const int inner = 1024;
    const int T = Nn >> 7;                    // 128-wide key tiles

    auto issue_kv = [&](int kt) {
        __bf16* Kc = Ks + (kt & 1) * (128 * 72);
        __bf16* Vc = Vs + (kt & 1) * (64 * 136);
        const int r  = tid >> 3;
        const int d8 = (tid & 7) << 3;
        #pragma unroll
        for (int i = 0; i < 4; ++i)
            async_b128(&Kc[(r + i * 32) * 72 + d8],
                       kv + (size_t)(b * Nn + kt * 128 + r + i * 32) * 2048 + h * 64 + d8);
        const int d  = tid >> 2;               // 0..63
        const int r8 = (tid & 3) << 3;         // 0/8/16/24
        #pragma unroll
        for (int i = 0; i < 4; ++i)
            async_b128(&Vc[d * 136 + r8 + i * 32],
                       vt + (size_t)(bh * 64 + d) * Nn + kt * 128 + r8 + i * 32);
    };

    // Prologue: Q tile + first K/V tile
    {
        const int r  = tid >> 3;
        const int d8 = (tid & 7) << 3;
        #pragma unroll
        for (int i = 0; i < 4; ++i)
            async_b128(&Qs[(r + i * 32) * 72 + d8],
                       q + (size_t)(b * Nn + qt * 128 + r + i * 32) * inner + h * 64 + d8);
    }
    issue_kv(0);
    if (tid < 128) { mrun[tid] = -1e30f; lrun[tid] = 0.f; }
    wait_async0();
    __syncthreads();

    // Hoist loop-invariant Q fragments (4 x 8 VGPRs)
    FragAB afrQ[2][2];
    #pragma unroll
    for (int ks2 = 0; ks2 < 2; ++ks2)
        #pragma unroll
        for (int mt = 0; mt < 2; ++mt) {
            int row = wm * 32 + mt * 16 + l16;
            int ko  = ks2 * 32 + half16 * 16;
            afrQ[ks2][mt].u[0] = *(const uint4*)&Qs[row * 72 + ko];
            afrQ[ks2][mt].u[1] = *(const uint4*)&Qs[row * 72 + ko + 8];
        }

    v8f accO[2][2] = {};
    FragAB afr, bfr;

    for (int kt = 0; kt < T; ++kt) {
        const __bf16* Kc = Ks + (kt & 1) * (128 * 72);
        const __bf16* Vc = Vs + (kt & 1) * (64 * 136);
        if (kt + 1 < T) issue_kv(kt + 1);     // DMA overlaps this tile's compute

        // S = Q * K^T  (M=128, N=128 keys, K=64 dh)
        v8f accS[2][4] = {};
        #pragma unroll
        for (int ks2 = 0; ks2 < 2; ++ks2) {
            #pragma unroll
            for (int mt = 0; mt < 2; ++mt) {
                #pragma unroll
                for (int nt = 0; nt < 4; ++nt) {
                    int col = wn * 64 + nt * 16 + l16;
                    int kb  = ks2 * 32 + half16 * 16;
                    bfr.u[0] = *(const uint4*)&Kc[col * 72 + kb];
                    bfr.u[1] = *(const uint4*)&Kc[col * 72 + kb + 8];
                    accS[mt][nt] = wmma_bf16(afrQ[ks2][mt], bfr, accS[mt][nt]);
                }
            }
        }

        // partial row-max over this wave's 64 columns
        #pragma unroll
        for (int mt = 0; mt < 2; ++mt)
            #pragma unroll
            for (int r = 0; r < 8; ++r) {
                float v = fmaxf(fmaxf(accS[mt][0][r], accS[mt][1][r]),
                                fmaxf(accS[mt][2][r], accS[mt][3][r]));
                v = fmaxf(v, __shfl_xor(v, 1));
                v = fmaxf(v, __shfl_xor(v, 2));
                v = fmaxf(v, __shfl_xor(v, 4));
                v = fmaxf(v, __shfl_xor(v, 8));
                if (l16 == 0) pmax[wn * 128 + wm * 32 + mt * 16 + half16 * 8 + r] = v;
            }
        __syncthreads();

        if (tid < 128) {
            float mnew = fmaxf(mrun[tid], fmaxf(pmax[tid], pmax[128 + tid]));
            float sc   = __expf(mrun[tid] - mnew);
            scl[tid]  = sc;
            mrun[tid] = mnew;
            lrun[tid] *= sc;
        }
        __syncthreads();

        // rescale O, P = exp(S - m) -> Ps (bf16), partial row sums
        #pragma unroll
        for (int mt = 0; mt < 2; ++mt)
            #pragma unroll
            for (int r = 0; r < 8; ++r) {
                int row = wm * 32 + mt * 16 + half16 * 8 + r;
                float m  = mrun[row];
                float sc = scl[row];
                accO[mt][0][r] *= sc;
                accO[mt][1][r] *= sc;
                float ps = 0.f;
                #pragma unroll
                for (int nt = 0; nt < 4; ++nt) {
                    float p = __expf(accS[mt][nt][r] - m);
                    Ps[row * 136 + wn * 64 + nt * 16 + l16] = (__bf16)p;
                    ps += p;
                }
                ps += __shfl_xor(ps, 1);
                ps += __shfl_xor(ps, 2);
                ps += __shfl_xor(ps, 4);
                ps += __shfl_xor(ps, 8);
                if (l16 == 0) psum[wn * 128 + row] = ps;
            }
        __syncthreads();
        if (tid < 128) lrun[tid] += psum[tid] + psum[128 + tid];

        // O += P * V  (M=128, N=64 dh, K=128 keys)
        #pragma unroll
        for (int ks = 0; ks < 128; ks += 32) {
            #pragma unroll
            for (int mt = 0; mt < 2; ++mt) {
                int row = wm * 32 + mt * 16 + l16;
                int ko  = ks + half16 * 16;
                afr.u[0] = *(const uint4*)&Ps[row * 136 + ko];
                afr.u[1] = *(const uint4*)&Ps[row * 136 + ko + 8];
                #pragma unroll
                for (int nt = 0; nt < 2; ++nt) {
                    int col = wn * 32 + nt * 16 + l16;
                    int kb  = ks + half16 * 16;
                    bfr.u[0] = *(const uint4*)&Vc[col * 136 + kb];
                    bfr.u[1] = *(const uint4*)&Vc[col * 136 + kb + 8];
                    accO[mt][nt] = wmma_bf16(afr, bfr, accO[mt][nt]);
                }
            }
        }

        wait_async0();        // next tile's DMA complete (this wave)
        __syncthreads();      // all waves done with this tile's LDS + DMA visible
    }

    // normalize and write bf16 output
    {
        __bf16* base = o + (size_t)(b * Nn + qt * 128 + wm * 32 + half16 * 8) * inner
                         + h * 64 + wn * 32 + l16;
        #pragma unroll
        for (int mt = 0; mt < 2; ++mt)
            #pragma unroll
            for (int r = 0; r < 8; ++r) {
                int row = wm * 32 + mt * 16 + half16 * 8 + r;
                float invl = 1.0f / lrun[row];
                __bf16* rp = base + (size_t)(mt * 16 + r) * inner;
                #pragma unroll
                for (int nt = 0; nt < 2; ++nt)
                    rp[nt * 16] = (__bf16)(accO[mt][nt][r] * invl);
            }
    }
}

// ---------------------------------------------------------------------------
// Host-side launcher
// ---------------------------------------------------------------------------
extern "C" void kernel_launch(void* const* d_in, const int* in_sizes, int n_in,
                              void* d_out, int out_size, void* d_ws, size_t ws_size,
                              hipStream_t stream) {
    (void)in_sizes; (void)n_in; (void)out_size; (void)ws_size;

    const float* x     = (const float*)d_in[0];
    const float* gamma = (const float*)d_in[1];
    const float* beta  = (const float*)d_in[2];
    const float* Wq    = (const float*)d_in[3];
    const float* Wkv   = (const float*)d_in[4];
    const float* Wout  = (const float*)d_in[5];
    float* out = (float*)d_out;

    const int B = 2, N = 2048, DIM = 1024, INNER = 1024;
    const int M = B * N;  // 4096

    char* ws = (char*)d_ws;
    __bf16* xn     = (__bf16*)ws; ws += (size_t)M * DIM * 2;
    __bf16* Wq_t   = (__bf16*)ws; ws += (size_t)DIM * INNER * 2;          // [INNER][DIM]
    __bf16* Wkv_t  = (__bf16*)ws; ws += (size_t)DIM * 2 * INNER * 2;      // [2*INNER][DIM]
    __bf16* Wout_t = (__bf16*)ws; ws += (size_t)INNER * DIM * 2;          // [DIM][INNER]
    __bf16* qb     = (__bf16*)ws; ws += (size_t)M * INNER * 2;
    __bf16* kvb    = (__bf16*)ws; ws += (size_t)M * 2 * INNER * 2;
    __bf16* ob     = (__bf16*)ws; ws += (size_t)M * INNER * 2;
    __bf16* vtb    = (__bf16*)ws; ws += (size_t)M * INNER * 2;            // [B*H*64][N]

    transpose_cast_bf16<<<dim3(INNER / 32, DIM / 32), 256, 0, stream>>>(Wq, Wq_t, DIM, INNER);
    transpose_cast_bf16<<<dim3(2 * INNER / 32, DIM / 32), 256, 0, stream>>>(Wkv, Wkv_t, DIM, 2 * INNER);
    transpose_cast_bf16<<<dim3(DIM / 32, INNER / 32), 256, 0, stream>>>(Wout, Wout_t, INNER, DIM);

    layernorm_bf16<<<M, 256, 0, stream>>>(x, gamma, beta, xn, DIM);

    gemm_bf16_wmma<false><<<dim3(INNER / 128, M / 128), 256, 0, stream>>>(
        xn, Wq_t, qb, M, DIM, INNER);
    gemm_bf16_wmma<false><<<dim3(2 * INNER / 128, M / 128), 256, 0, stream>>>(
        xn, Wkv_t, kvb, M, DIM, 2 * INNER);

    build_vt<<<dim3(N / 64, B * 16), 256, 0, stream>>>(kvb, vtb, N);

    size_t smem = (size_t)(128 * 72 + 2 * 128 * 72 + 2 * 64 * 136 + 128 * 136) * 2
                + (size_t)(128 * 3 + 256 * 2) * 4;
    attn_wmma<<<dim3(N / 128, 16, B), 256, smem, stream>>>(qb, kvb, vtb, ob, N);

    gemm_bf16_wmma<true><<<dim3(DIM / 128, M / 128), 256, 0, stream>>>(
        ob, Wout_t, out, M, INNER, DIM);
}
